// JambaMoE_9156870275175
// MI455X (gfx1250) — compile-verified, matrix operand order checked by reading, not run on hardware
//
#include <hip/hip_runtime.h>
#include <hip/hip_bf16.h>
#include <cstdint>

// Problem constants (JambaMoE): B=2, S=4096, D=2048, E=8, I=4096, TOP_K=2
#define D_   2048
#define E_   8
#define I_   4096
#define T_   8192
#define TOPK_ 2

typedef __attribute__((ext_vector_type(16))) __bf16 v16bf;
typedef __attribute__((ext_vector_type(8)))  __bf16 v8bf;
typedef __attribute__((ext_vector_type(4)))  __bf16 v4bf;
typedef __attribute__((ext_vector_type(8)))  float  v8f;
typedef uint32_t u32;
typedef __attribute__((ext_vector_type(4))) u32 u32x4;
typedef __attribute__((ext_vector_type(8))) u32 u32x8;

__device__ __forceinline__ v8f wmma_bf16(v16bf a, v16bf b, v8f c) {
  return __builtin_amdgcn_wmma_f32_16x16x32_bf16(false, a, false, b, (short)0, c,
                                                 false, false);
}

// CDNA5 async global->LDS copy, 16B per lane (ASYNCcnt-tracked, VGPR-bypass).
__device__ __forceinline__ void async_ld16(uint32_t lds_addr, const void* gptr) {
  asm volatile("global_load_async_to_lds_b128 %0, %1, off"
               :: "v"(lds_addr), "v"(gptr) : "memory");
}
#define WAIT_ASYNC(n) asm volatile("s_wait_asynccnt " #n ::: "memory")

__device__ __forceinline__ uint32_t lds_addr_of(const void* p) {
  // Flat shared pointers carry the wave-relative LDS offset in the low 32 bits.
  return (uint32_t)(uintptr_t)p;
}

// ---- Tensor Data Mover: 2D strided tile -> contiguous LDS (TENSORcnt) ------
// D# group1 (static per kernel): data_size=2B, tensor dims, tile dims, stride.
__device__ __forceinline__ u32x8 tdm_desc_2d(u32 tensor_d0, u32 tensor_d1,
                                             u32 tile_d0, u32 tile_d1,
                                             u32 stride_elems) {
  u32x8 g;
  g[0] = 0x00010000u;                                   // data_size=1 (2 bytes)
  g[1] = (tensor_d0 & 0xFFFFu) << 16;                   // tensor_dim0[15:0]
  g[2] = ((tensor_d1 & 0xFFFFu) << 16) | (tensor_d0 >> 16);
  g[3] = (tile_d0 << 16) | (tensor_d1 >> 16);           // tile_dim0
  g[4] = tile_d1;                                       // tile_dim1 (dim2=0)
  g[5] = stride_elems;                                  // tensor_dim0_stride lo
  g[6] = 0u;                                            // stride hi / dim1_stride
  g[7] = 0u;
  return g;
}
// D# group0 (per call): count=1, lds_addr, 57-bit global_addr, type=2.
__device__ __forceinline__ void tdm_load_2d(u32 lds_addr, const void* gaddr,
                                            u32x8 g1) {
  uint64_t ga = (uint64_t)(uintptr_t)gaddr;
  u32x4 g0;
  g0[0] = 1u;
  g0[1] = lds_addr;
  g0[2] = (u32)ga;
  g0[3] = ((u32)(ga >> 32) & 0x01FFFFFFu) | 0x80000000u;  // type=2 ("image")
  asm volatile("tensor_load_to_lds %0, %1" :: "s"(g0), "s"(g1) : "memory");
}

// Single-instruction fp32 atomic add (no-return form, STOREcnt-tracked).
__device__ __forceinline__ void atomic_add_f32(float* p, float v) {
  asm volatile("global_atomic_add_f32 %0, %1, off" :: "v"(p), "v"(v) : "memory");
}

// 16x32 bf16 fragment from an LDS tile (row-major, stride 32 elements).
__device__ __forceinline__ v16bf load_frag(const __bf16* base, int row, int hi) {
  v8bf lo = *(const v8bf*)(base + row * 32 + hi * 8);
  v8bf hh = *(const v8bf*)(base + row * 32 + 16 + hi * 8);
  return __builtin_shufflevector(lo, hh, 0, 1, 2, 3, 4, 5, 6, 7,
                                 8, 9, 10, 11, 12, 13, 14, 15);
}

__device__ __forceinline__ v8f v8f_zero() {
  v8f z;
#pragma unroll
  for (int q = 0; q < 8; ++q) z[q] = 0.f;
  return z;
}

// ---------------------------------------------------------------------------
// fp32 -> bf16 bulk convert (vectorized x4)
// ---------------------------------------------------------------------------
__global__ __launch_bounds__(256) void convert_f32_bf16(
    const float* __restrict__ src, __bf16* __restrict__ dst, int n4) {
  int i = blockIdx.x * 256 + threadIdx.x;
  if (i >= n4) return;
  float4 v = ((const float4*)src)[i];
  v4bf o;
  o[0] = (__bf16)v.x; o[1] = (__bf16)v.y; o[2] = (__bf16)v.z; o[3] = (__bf16)v.w;
  ((v4bf*)dst)[i] = o;
}

// ---------------------------------------------------------------------------
// Router: one wave per token.  logits = h @ rw^T (E=8), softmax, top-2.
// ---------------------------------------------------------------------------
__global__ __launch_bounds__(256) void router_kernel(
    const float* __restrict__ h, const float* __restrict__ rw,
    int* __restrict__ topk_id, float* __restrict__ topk_w) {
  __shared__ __align__(16) float rws[E_ * D_];
  const int tid = threadIdx.x;
  for (int i = tid; i < E_ * D_ / 4; i += 256)
    ((float4*)rws)[i] = ((const float4*)rw)[i];
  __syncthreads();

  const int lane = tid & 31;
  const int t = blockIdx.x * 8 + (tid >> 5);
  const float* hp = h + (size_t)t * D_;

  float acc[E_];
#pragma unroll
  for (int e = 0; e < E_; ++e) acc[e] = 0.f;
  for (int j = lane; j < D_; j += 32) {
    float hv = hp[j];
#pragma unroll
    for (int e = 0; e < E_; ++e) acc[e] += hv * rws[e * D_ + j];
  }
#pragma unroll
  for (int e = 0; e < E_; ++e) {
#pragma unroll
    for (int off = 16; off >= 1; off >>= 1)
      acc[e] += __shfl_xor(acc[e], off, 32);
  }
  if (lane == 0) {
    float mx = acc[0];
#pragma unroll
    for (int e = 1; e < E_; ++e) mx = fmaxf(mx, acc[e]);
    float p[E_], s = 0.f;
#pragma unroll
    for (int e = 0; e < E_; ++e) { p[e] = __expf(acc[e] - mx); s += p[e]; }
    float inv = __builtin_amdgcn_rcpf(s);
    int i0 = 0;
#pragma unroll
    for (int e = 1; e < E_; ++e) if (p[e] > p[i0]) i0 = e;
    int i1 = (i0 == 0) ? 1 : 0;
#pragma unroll
    for (int e = 0; e < E_; ++e) if (e != i0 && p[e] > p[i1]) i1 = e;
    topk_id[t * 2 + 0] = i0;
    topk_id[t * 2 + 1] = i1;
    topk_w[t * 2 + 0] = p[i0] * inv;
    topk_w[t * 2 + 1] = p[i1] * inv;
  }
}

// ---------------------------------------------------------------------------
// Deterministic compaction: wave e builds expert e's ordered token list.
// ---------------------------------------------------------------------------
__global__ __launch_bounds__(256) void compact_kernel(
    const int* __restrict__ topk_id, const float* __restrict__ topk_w,
    int* __restrict__ tok_ids, float* __restrict__ tok_w,
    int* __restrict__ counts, int* __restrict__ offsets) {
  const int tid = threadIdx.x;
  const int e = tid >> 5;
  const int lane = tid & 31;
  int cnt = 0;
  for (int base0 = 0; base0 < T_; base0 += 32) {
    int t = base0 + lane;
    int id0 = topk_id[t * 2 + 0];
    int id1 = topk_id[t * 2 + 1];
    int slot = (id0 == e) ? 0 : ((id1 == e) ? 1 : -1);
    unsigned mask = (unsigned)(__ballot(slot >= 0) & 0xffffffffull);
    if (slot >= 0) {
      int pos = cnt + __popc(mask & ((1u << lane) - 1u));
      tok_ids[e * T_ + pos] = t;
      tok_w[e * T_ + pos] = topk_w[t * 2 + slot];
    }
    cnt += __popc(mask);
  }
  if (lane == 0) counts[e] = cnt;
  __syncthreads();
  if (tid == 0) {
    int o = 0;
#pragma unroll
    for (int i = 0; i < E_; ++i) { offsets[i] = o; o += counts[i]; }
  }
}

// ---------------------------------------------------------------------------
// GEMM1 + SiLU: act = silu(h@Wg^T) * (h@Wu^T), bf16 out.
// Tile 64 tokens x 256 cols (+ up), 8 waves (2x4) of 32x64.  3-stage pipe:
// B tiles via TDM (wave 0, 2 tensor ops/stage), gathered A rows via per-lane
// async loads (1/thread/stage).
// ---------------------------------------------------------------------------
__global__ __launch_bounds__(256) void gemm1_silu_kernel(
    const __bf16* __restrict__ h_bf, const __bf16* __restrict__ ws_bf,
    const int* __restrict__ tok_ids, const int* __restrict__ counts,
    const int* __restrict__ offsets, __bf16* __restrict__ act) {
  const int e = blockIdx.z;
  const int cnt = counts[e];
  const int mt = blockIdx.y * 64;
  if (mt >= cnt) return;
  const int nb = blockIdx.x * 256;

  __shared__ __align__(16) __bf16 As[3][64 * 32];    // 12 KB
  __shared__ __align__(16) __bf16 Bg[3][256 * 32];   // 48 KB
  __shared__ __align__(16) __bf16 Bu[3][256 * 32];   // 48 KB
  __shared__ int tok_s[64];

  const int tid = threadIdx.x;
  if (tid < 64) {
    int j = mt + tid;
    tok_s[tid] = (j < cnt) ? tok_ids[e * T_ + j] : 0;
  }
  __syncthreads();

  const int wid = tid >> 5, lane = tid & 31;
  const int wm = wid >> 2, wn = wid & 3;    // 2 x 4 wave grid (32m x 64n each)
  const int hi = lane >> 4, lm = lane & 15;

  const int arow = tid >> 2, ach = tid & 3;           // A: 64 rows x 4 chunks
  const __bf16* a_src = h_bf + (size_t)tok_s[arow] * D_ + ach * 8;
  const size_t wbase = (size_t)e * (size_t)(2 * I_) * D_;
  const __bf16* bg_base = ws_bf + wbase + (size_t)nb * D_;
  const __bf16* bu_base = ws_bf + wbase + (size_t)(I_ + nb) * D_;
  const u32x8 descB = tdm_desc_2d(D_, 2 * I_, 32, 256, D_);

  auto stage_a = [&](int buf, int k0) {
    async_ld16(lds_addr_of(&As[buf][arow * 32 + ach * 8]), a_src + k0);
  };
  auto stage_b = [&](int buf, int k0) {
    tdm_load_2d(lds_addr_of(&Bg[buf][0]), bg_base + k0, descB);
    tdm_load_2d(lds_addr_of(&Bu[buf][0]), bu_base + k0, descB);
  };

  v8f cg[2][4], cu[2][4];
#pragma unroll
  for (int i = 0; i < 2; ++i)
#pragma unroll
    for (int j = 0; j < 4; ++j) { cg[i][j] = v8f_zero(); cu[i][j] = v8f_zero(); }

  stage_a(0, 0);
  stage_a(1, 32);
  if (wid == 0) { stage_b(0, 0); stage_b(1, 32); }

  int it = 0;
  for (int k0 = 0; k0 < D_; k0 += 32, ++it) {
    const int buf = it % 3;
    if (k0 + 32 < D_) { WAIT_ASYNC(1); } else { WAIT_ASYNC(0); }  // my A chunk
    if (wid == 0) {                                   // oldest stage's B tiles
      if (k0 + 32 < D_) __builtin_amdgcn_s_wait_tensorcnt(2);
      else              __builtin_amdgcn_s_wait_tensorcnt(0);
    }
    __syncthreads();
    if (k0 + 64 < D_) {
      stage_a((it + 2) % 3, k0 + 64);
      if (wid == 0) stage_b((it + 2) % 3, k0 + 64);
    }

    v16bf bg[4], bu[4];
#pragma unroll
    for (int j = 0; j < 4; ++j) {
      int br = wn * 64 + j * 16 + lm;
      bg[j] = load_frag(Bg[buf], br, hi);
      bu[j] = load_frag(Bu[buf], br, hi);
    }
#pragma unroll
    for (int i = 0; i < 2; ++i) {
      v16bf a = load_frag(As[buf], wm * 32 + i * 16 + lm, hi);
#pragma unroll
      for (int j = 0; j < 4; ++j) {
        cg[i][j] = wmma_bf16(a, bg[j], cg[i][j]);
        cu[i][j] = wmma_bf16(a, bu[j], cu[i][j]);
      }
    }
  }

  const int rowbase = offsets[e] + mt;
#pragma unroll
  for (int i = 0; i < 2; ++i)
#pragma unroll
    for (int j = 0; j < 4; ++j)
#pragma unroll
      for (int r = 0; r < 8; ++r) {
        int m = wm * 32 + i * 16 + hi * 8 + r;
        if (mt + m < cnt) {
          int n = nb + wn * 64 + j * 16 + lm;
          float g = cg[i][j][r], u = cu[i][j][r];
          float s = g * __builtin_amdgcn_rcpf(1.f + __expf(-g)) * u;
          act[(size_t)(rowbase + m) * I_ + n] = (__bf16)s;
        }
      }
}

// ---------------------------------------------------------------------------
// GEMM2 (down proj): out[tok, d] += w * (act_row @ w2[e]^T).
// Tile 64 rows x 256 d-cols.  Both A (contiguous routed rows) and B are
// regular 2D tiles -> fully TDM-staged (wave 0, 2 tensor ops/stage).
// ---------------------------------------------------------------------------
__global__ __launch_bounds__(256) void gemm2_down_kernel(
    const __bf16* __restrict__ act, const __bf16* __restrict__ w2_bf,
    const int* __restrict__ tok_ids, const float* __restrict__ tok_w,
    const int* __restrict__ counts, const int* __restrict__ offsets,
    float* __restrict__ out) {
  const int e = blockIdx.z;
  const int cnt = counts[e];
  const int mt = blockIdx.y * 64;
  if (mt >= cnt) return;
  const int nb = blockIdx.x * 256;

  __shared__ __align__(16) __bf16 As[3][64 * 32];    // 12 KB
  __shared__ __align__(16) __bf16 Bs[3][256 * 32];   // 48 KB
  __shared__ int tok_s[64];
  __shared__ float w_s[64];

  const int tid = threadIdx.x;
  if (tid < 64) {
    int j = mt + tid;
    tok_s[tid] = (j < cnt) ? tok_ids[e * T_ + j] : 0;
    w_s[tid] = (j < cnt) ? tok_w[e * T_ + j] : 0.f;
  }
  __syncthreads();

  const int wid = tid >> 5, lane = tid & 31;
  const int wm = wid >> 2, wn = wid & 3;
  const int hi = lane >> 4, lm = lane & 15;

  const int rowbase = offsets[e] + mt;
  const size_t w2base = (size_t)e * (size_t)D_ * I_;
  const __bf16* a_base = act + (size_t)rowbase * I_;
  const __bf16* b_base = w2_bf + w2base + (size_t)nb * I_;
  const u32x8 descA = tdm_desc_2d(I_, (u32)(T_ * TOPK_ + 64), 32, 64, I_);
  const u32x8 descB = tdm_desc_2d(I_, D_, 32, 256, I_);

  auto stage = [&](int buf, int k0) {
    tdm_load_2d(lds_addr_of(&As[buf][0]), a_base + k0, descA);
    tdm_load_2d(lds_addr_of(&Bs[buf][0]), b_base + k0, descB);
  };

  v8f c[2][4];
#pragma unroll
  for (int i = 0; i < 2; ++i)
#pragma unroll
    for (int j = 0; j < 4; ++j) c[i][j] = v8f_zero();

  if (wid == 0) { stage(0, 0); stage(1, 32); }

  int it = 0;
  for (int k0 = 0; k0 < I_; k0 += 32, ++it) {
    const int buf = it % 3;
    if (wid == 0) {
      if (k0 + 32 < I_) __builtin_amdgcn_s_wait_tensorcnt(2);
      else              __builtin_amdgcn_s_wait_tensorcnt(0);
    }
    __syncthreads();
    if (k0 + 64 < I_ && wid == 0) stage((it + 2) % 3, k0 + 64);

    v16bf bf2[4];
#pragma unroll
    for (int j = 0; j < 4; ++j)
      bf2[j] = load_frag(Bs[buf], wn * 64 + j * 16 + lm, hi);
#pragma unroll
    for (int i = 0; i < 2; ++i) {
      v16bf a = load_frag(As[buf], wm * 32 + i * 16 + lm, hi);
#pragma unroll
      for (int j = 0; j < 4; ++j) c[i][j] = wmma_bf16(a, bf2[j], c[i][j]);
    }
  }

#pragma unroll
  for (int i = 0; i < 2; ++i)
#pragma unroll
    for (int j = 0; j < 4; ++j)
#pragma unroll
      for (int r = 0; r < 8; ++r) {
        int m = wm * 32 + i * 16 + hi * 8 + r;
        if (mt + m < cnt) {
          int n = nb + wn * 64 + j * 16 + lm;
          atomic_add_f32(&out[(size_t)tok_s[m] * D_ + n], c[i][j][r] * w_s[m]);
        }
      }
}

// ---------------------------------------------------------------------------
// Host launcher
// ---------------------------------------------------------------------------
extern "C" void kernel_launch(void* const* d_in, const int* in_sizes, int n_in,
                              void* d_out, int out_size, void* d_ws, size_t ws_size,
                              hipStream_t stream) {
  (void)in_sizes; (void)n_in; (void)ws_size;
  const float* h   = (const float*)d_in[0];   // [T, D] fp32
  const float* rw  = (const float*)d_in[1];   // [E, D] fp32
  const float* wsp = (const float*)d_in[2];   // [E, 2I, D] fp32
  const float* w2p = (const float*)d_in[3];   // [E, D, I] fp32
  float* out = (float*)d_out;                 // [T, D] fp32

  char* base = (char*)d_ws;
  size_t off = 0;
  auto alloc = [&](size_t bytes) -> void* {
    void* p = base + off;
    off = (off + bytes + 255) & ~(size_t)255;
    return p;
  };
  int*    counts  = (int*)   alloc((size_t)E_ * 4);
  int*    offsets = (int*)   alloc((size_t)E_ * 4);
  int*    topk_id = (int*)   alloc((size_t)T_ * 2 * 4);
  float*  topk_w  = (float*) alloc((size_t)T_ * 2 * 4);
  int*    tok_ids = (int*)   alloc((size_t)E_ * T_ * 4);
  float*  tok_w   = (float*) alloc((size_t)E_ * T_ * 4);
  __bf16* h_bf    = (__bf16*)alloc((size_t)T_ * D_ * 2);
  __bf16* ws_bf   = (__bf16*)alloc((size_t)E_ * 2 * I_ * D_ * 2);
  __bf16* w2_bf   = (__bf16*)alloc((size_t)E_ * D_ * I_ * 2);
  __bf16* act     = (__bf16*)alloc(((size_t)T_ * TOPK_ + 64) * I_ * 2);

  hipMemsetAsync(out, 0, (size_t)out_size * sizeof(float), stream);

  int n4h = T_ * (D_ / 4);
  convert_f32_bf16<<<(n4h + 255) / 256, 256, 0, stream>>>(h, h_bf, n4h);
  int n4w = E_ * 2 * I_ * (D_ / 4);
  convert_f32_bf16<<<(n4w + 255) / 256, 256, 0, stream>>>(wsp, ws_bf, n4w);
  int n4w2 = E_ * D_ * (I_ / 4);
  convert_f32_bf16<<<(n4w2 + 255) / 256, 256, 0, stream>>>(w2p, w2_bf, n4w2);

  router_kernel<<<T_ / 8, 256, 0, stream>>>(h, rw, topk_id, topk_w);
  compact_kernel<<<1, 256, 0, stream>>>(topk_id, topk_w, tok_ids, tok_w,
                                        counts, offsets);
  gemm1_silu_kernel<<<dim3(I_ / 256, T_ / 64, E_), 256, 0, stream>>>(
      h_bf, ws_bf, tok_ids, counts, offsets, act);
  gemm2_down_kernel<<<dim3(D_ / 256, T_ / 64, E_), 256, 0, stream>>>(
      act, w2_bf, tok_ids, tok_w, counts, offsets, out);
}
